// TrajectoryEncoder_40295383171617
// MI455X (gfx1250) — matrix operand order repaired and checked by reading, not compiled, not run on hardware
//
#include <hip/hip_runtime.h>
#include <hip/hip_bf16.h>

typedef _Float16 half_t;
typedef __attribute__((ext_vector_type(16))) _Float16 v16h;
typedef __attribute__((ext_vector_type(8)))  _Float16 v8h;
typedef __attribute__((ext_vector_type(8)))  float    v8f;

constexpr int kB   = 8192;   // batch
constexpr int kT   = 32;     // timesteps
constexpr int kEmb = 128;    // embedding width (layer-0 input)
constexpr int kH   = 256;    // hidden
constexpr int kG   = 1024;   // 4*H gate width

// ---------------------------------------------------------------------------
// Weight prep: combined transposed fp16 weight  Wt[N=1024][K=din+H]
// from fp32 W [din][1024] and U [H][1024].
// ---------------------------------------------------------------------------
__global__ void prep_weights_kernel(const float* __restrict__ W,
                                    const float* __restrict__ U,
                                    half_t* __restrict__ Wt,
                                    int din) {
  const int ktot = din + kH;
  const int idx = blockIdx.x * blockDim.x + threadIdx.x;
  const int total = kG * ktot;
  if (idx >= total) return;
  const int n = idx / ktot;
  const int k = idx - n * ktot;
  const float v = (k < din) ? W[(size_t)k * kG + n]
                            : U[(size_t)(k - din) * kG + n];
  Wt[(size_t)n * ktot + k] = (half_t)v;
}

// Branch-free transcendentals -> v_exp_f32 / v_rcp_f32 (TRANS ops, no EXEC
// divergence inside the hot loop).
__device__ __forceinline__ float fsigmoid_(float x) {
  return __builtin_amdgcn_rcpf(
      1.0f + __builtin_amdgcn_exp2f(x * -1.44269504088896340736f));
}
__device__ __forceinline__ float ftanh_(float x) {
  const float s = __builtin_amdgcn_rcpf(
      1.0f + __builtin_amdgcn_exp2f(x * -2.88539008177792681472f));
  return fmaf(2.0f, s, -1.0f);
}

// ---------------------------------------------------------------------------
// Fused LSTM layer.  One workgroup = one 16-row batch tile, all T timesteps.
//  - h (fp16) persists in LDS; c persists in REGISTERS (gate-aligned N-tiling)
//  - per step: z = [x_t , h] @ Wt^T + b via v_wmma_f32_16x16x32_f16
//  - wave w computes ALL 4 gates for hidden columns [w*32, w*32+32), so the
//    LSTM cell update is register-resident (no z round-trip through LDS)
//  - x_{t+1} is register-prefetched under the GEMM
// ---------------------------------------------------------------------------
template <int DIN, bool FIRST, bool LAST>
__global__ void __launch_bounds__(256)
lstm_layer_kernel(const float* __restrict__ traj,   // [B,T,2]   (FIRST)
                  const float* __restrict__ W_emb,  // [2,128]   (FIRST)
                  const float* __restrict__ b_emb,  // [128]     (FIRST)
                  const half_t* __restrict__ Xin,   // [B*T,DIN] (!FIRST)
                  const half_t* __restrict__ Wt,    // [1024][DIN+H]
                  const float* __restrict__ bias,   // [1024]
                  float* __restrict__ Yf,           // [B,T,H] fp32 (LAST)
                  half_t* __restrict__ Yh,          // [B,T,H] fp16 (!LAST)
                  float* __restrict__ h_out,        // [B,H]
                  float* __restrict__ c_out) {      // [B,H]
  constexpr int KTOT = DIN + kH;
  constexpr int KF   = KTOT / 32;

  extern __shared__ char smem[];
  half_t* Atile = (half_t*)smem;   // [16][KTOT]  ([x_t , h])

  const int tid   = threadIdx.x;
  const int lane  = tid & 31;
  const int wave  = tid >> 5;          // 0..7 (wave32)
  const int row16 = lane & 15;
  const int khalf = lane >> 4;         // 0 or 1
  const int row0  = blockIdx.x * 16;   // batch-tile origin

  // ---- init: h = 0 in LDS (c is register-resident)
  {
    const int r = tid >> 4, c0 = (tid & 15) * 16;
#pragma unroll
    for (int i = 0; i < 16; ++i) Atile[r * KTOT + DIN + c0 + i] = (half_t)0.0f;
  }

  // ---- per-wave constants.  N-tile nt -> physical gate column
  //   col = (nt>>1)*kH + wave*32 + (nt&1)*16 + row16    (gate order i,f,g,o)
  // All four gates share one (VGPR,lane)<->(M,j) mapping.  A single per-lane
  // base pointer + compile-time byte offsets lets the loads use IOFFSET.
  float bb[8];
#pragma unroll
  for (int nt = 0; nt < 8; ++nt)
    bb[nt] = bias[(nt >> 1) * kH + wave * 32 + (nt & 1) * 16 + row16];
  const half_t* bbase =
      Wt + (size_t)(wave * 32 + row16) * KTOT + khalf * 16;
  const int jlane = wave * 32 + row16;   // hidden column of sub-slice p=0

  // register-resident cell state: c(M = v+8*khalf, j = jlane + p*16)
  float creg[16];
#pragma unroll
  for (int i = 0; i < 16; ++i) creg[i] = 0.0f;

  // ---- register-staged x_t (prefetched one step ahead)
  const int r  = tid >> 4;         // staging row 0..15
  const int cc = (tid & 15);       // staging column group
  v8h  xr0 = {}, xr1 = {};
  float e_in0 = 0.0f, e_in1 = 0.0f;
  float ew0[8], ew1[8], eb[8];
  if constexpr (FIRST) {
    const int e0 = cc * 8;
#pragma unroll
    for (int i = 0; i < 8; ++i) {       // loop-invariant embed weights
      ew0[i] = W_emb[e0 + i];
      ew1[i] = W_emb[kEmb + e0 + i];
      eb[i]  = b_emb[e0 + i];
    }
    const float* tp = traj + (size_t)(row0 + r) * kT * 2;   // t = 0
    e_in0 = tp[0];
    e_in1 = tp[1];
  } else {
    const v8h* src = (const v8h*)(Xin + (size_t)(row0 + r) * kT * DIN + cc * 16);
    xr0 = src[0];
    xr1 = src[1];
  }

  __syncthreads();

#pragma unroll 1
  for (int t = 0; t < kT; ++t) {
    // ---- commit staged x_t into Atile[:, 0:DIN]
    if constexpr (FIRST) {
      const int e0 = cc * 8;
#pragma unroll
      for (int i = 0; i < 8; ++i) {
        const float v = fmaf(e_in0, ew0[i], fmaf(e_in1, ew1[i], eb[i]));
        Atile[r * KTOT + e0 + i] = (half_t)fmaxf(v, 0.0f);
      }
    } else {
      v8h* dst = (v8h*)(Atile + r * KTOT + cc * 16);
      dst[0] = xr0;
      dst[1] = xr1;
    }
    __syncthreads();

    // ---- prefetch x_{t+1} into registers (latency hides under the GEMM)
    {
      const int tn = (t + 1) & (kT - 1);   // wraps harmlessly at t = kT-1
      if constexpr (FIRST) {
        const float* tp = traj + ((size_t)(row0 + r) * kT + tn) * 2;
        e_in0 = tp[0];
        e_in1 = tp[1];
      } else {
        const v8h* src =
            (const v8h*)(Xin + ((size_t)(row0 + r) * kT + tn) * DIN + cc * 16);
        xr0 = src[0];
        xr1 = src[1];
      }
    }

    // ---- z = [x_t , h] @ Wt^T + b : 8 independent 16x16 accumulators/wave
    v8f acc[8];
#pragma unroll
    for (int nt = 0; nt < 8; ++nt) {
#pragma unroll
      for (int i = 0; i < 8; ++i) acc[nt][i] = bb[nt];
    }
#pragma unroll 1
    for (int kf = 0; kf < KF; ++kf) {
      // A fragment (contiguous k order, same as B: lane<16 -> K{0..15}, lane>=16 -> K{16..31})
      const half_t* ab = Atile + row16 * KTOT + kf * 32 + khalf * 16;
      const v8h alo = *(const v8h*)(ab);
      const v8h ahi = *(const v8h*)(ab + 8);
      const v16h a = __builtin_shufflevector(alo, ahi,
          0, 1, 2, 3, 4, 5, 6, 7, 8, 9, 10, 11, 12, 13, 14, 15);
      const half_t* bk = bbase + kf * 32;
#pragma unroll
      for (int nt = 0; nt < 8; ++nt) {
        // compile-time offset -> folds into the global_load IOFFSET
        const size_t boff = (size_t)((nt >> 1) * kH + (nt & 1) * 16) * KTOT;
        const v16h b = *(const v16h*)(bk + boff);
        acc[nt] = __builtin_amdgcn_wmma_f32_16x16x32_f16(
            false, a, false, b, (short)0, acc[nt], false, false);
      }
    }
    __syncthreads();   // all A reads done before h is overwritten below

    // ---- register-resident LSTM cell update (gate order i,f,g,o)
#pragma unroll
    for (int p = 0; p < 2; ++p) {
#pragma unroll
      for (int v = 0; v < 8; ++v) {
        const float zi = acc[0 + p][v];
        const float zf = acc[2 + p][v];
        const float zg = acc[4 + p][v];
        const float zo = acc[6 + p][v];
        const float cv = fsigmoid_(zf) * creg[p * 8 + v]
                       + fsigmoid_(zi) * ftanh_(zg);
        creg[p * 8 + v] = cv;
        const float hv = fsigmoid_(zo) * ftanh_(cv);
        const int M = v + 8 * khalf;          // row within 16-row tile
        const int j = jlane + p * 16;         // hidden column
        Atile[M * KTOT + DIN + j] = (half_t)hv;   // republish h for next step
        const size_t oidx = ((size_t)(row0 + M) * kT + t) * kH + j;
        if constexpr (LAST) Yf[oidx] = hv;
        else                Yh[oidx] = (half_t)hv;
        if (t == kT - 1) {
          h_out[(size_t)(row0 + M) * kH + j] = hv;
          c_out[(size_t)(row0 + M) * kH + j] = cv;
        }
      }
    }
    // next iteration's __syncthreads (after staging commit) orders these
    // h-writes against the next GEMM's A reads.
  }
}

// ---------------------------------------------------------------------------
extern "C" void kernel_launch(void* const* d_in, const int* in_sizes, int n_in,
                              void* d_out, int out_size, void* d_ws, size_t ws_size,
                              hipStream_t stream) {
  (void)in_sizes; (void)n_in; (void)out_size; (void)ws_size;
  const float* traj  = (const float*)d_in[0];
  const float* W_emb = (const float*)d_in[1];
  const float* b_emb = (const float*)d_in[2];
  const float* W0    = (const float*)d_in[3];
  const float* U0    = (const float*)d_in[4];
  const float* b0    = (const float*)d_in[5];
  const float* W1    = (const float*)d_in[6];
  const float* U1    = (const float*)d_in[7];
  const float* b1    = (const float*)d_in[8];

  float* out = (float*)d_out;
  float* y1  = out;                                       // [B,T,H]
  float* h0o = y1 + (size_t)kB * kT * kH;                 // [B,H]
  float* c0o = h0o + (size_t)kB * kH;
  float* h1o = c0o + (size_t)kB * kH;
  float* c1o = h1o + (size_t)kB * kH;

  char* ws = (char*)d_ws;
  const size_t wt0_bytes = (size_t)kG * (kEmb + kH) * sizeof(half_t);  // 768 KB
  const size_t wt1_bytes = (size_t)kG * (kH + kH) * sizeof(half_t);    // 1 MB
  half_t* Wt0 = (half_t*)ws;
  half_t* Wt1 = (half_t*)(ws + wt0_bytes);
  half_t* Y0  = (half_t*)(ws + wt0_bytes + wt1_bytes);                 // [B*T,256] fp16

  // --- weight prep (transposed fp16, combined [W;U])
  {
    const int tot0 = kG * (kEmb + kH);
    const int tot1 = kG * (kH + kH);
    prep_weights_kernel<<<(tot0 + 255) / 256, 256, 0, stream>>>(W0, U0, Wt0, kEmb);
    prep_weights_kernel<<<(tot1 + 255) / 256, 256, 0, stream>>>(W1, U1, Wt1, kH);
  }

  // --- fused layers (dynamic LDS: A-tile only)
  const size_t lds0 = (size_t)16 * (kEmb + kH) * sizeof(half_t);  // 12 KB
  const size_t lds1 = (size_t)16 * (kH + kH) * sizeof(half_t);    // 16 KB

  lstm_layer_kernel<kEmb, true, false>
      <<<kB / 16, 256, lds0, stream>>>(traj, W_emb, b_emb, nullptr,
                                       Wt0, b0, nullptr, Y0, h0o, c0o);
  lstm_layer_kernel<kH, false, true>
      <<<kB / 16, 256, lds1, stream>>>(nullptr, nullptr, nullptr, Y0,
                                       Wt1, b1, y1, nullptr, h1o, c1o);
}
